// MultiHeadAttention_60017872994776
// MI455X (gfx1250) — compile-verified, matrix operand order checked by reading, not run on hardware
//
#include <hip/hip_runtime.h>
#include <hip/hip_bf16.h>

// MHA forward for MI455X (gfx1250).
// WMMA f16 GEMMs, software-pipelined fragment loads, async K staging to LDS.
constexpr int H  = 16;
constexpr int D  = 1024;
constexpr int DH = 64;
constexpr int B  = 4;
constexpr int S  = 2048;

typedef __attribute__((ext_vector_type(16))) _Float16 v16h;
typedef __attribute__((ext_vector_type(8)))  _Float16 v8h;
typedef __attribute__((ext_vector_type(4)))  _Float16 v4h;
typedef __attribute__((ext_vector_type(8)))  float    v8f;

__device__ __forceinline__ v8f wmma_f16(v16h a, v16h b, v8f c) {
  return __builtin_amdgcn_wmma_f32_16x16x32_f16(false, a, false, b, (short)0, c,
                                                false, false);
}

// Fragment (ISA 7.12.2 layout): lane L -> row/col L%16, kb=(L/16)*8,
// elems {kb..kb+7, kb+16..kb+23}: two contiguous 16B loads (b128 / ds_b128).
__device__ __forceinline__ v16h load_frag(const _Float16* p) {
  v8h lo = *(const v8h*)p;
  v8h hi = *(const v8h*)(p + 16);
  v16h r;
#pragma unroll
  for (int i = 0; i < 8; ++i) { r[i] = lo[i]; r[8 + i] = hi[i]; }
  return r;
}

// Async-stage one 16x64-half K tile (2KB) into LDS. Per-lane src/dst carry
// row = lane%16 and 16B sub-chunk = (lane/16)*16B; the 4 instructions at
// offsets 0/32/64/96 cover the full 128B row (offset applies to both sides).
// Tracked by ASYNCcnt (+4 per call).
__device__ __forceinline__ void async_stage_tile(const _Float16* gsrc,
                                                 const _Float16* lds_dst) {
  const unsigned l = (unsigned)(size_t)(const void*)lds_dst; // LDS byte offset
  asm volatile(
      "global_load_async_to_lds_b128 %0, %1, off\n\t"
      "global_load_async_to_lds_b128 %0, %1, off offset:32\n\t"
      "global_load_async_to_lds_b128 %0, %1, off offset:64\n\t"
      "global_load_async_to_lds_b128 %0, %1, off offset:96"
      :: "v"(l), "v"(gsrc) : "memory");
}

// ---- staging kernels (run once; negligible vs the 1.07GB weights store) ----
__global__ void __launch_bounds__(256) cvt16_kernel(const float* __restrict__ X,
                                                    _Float16* __restrict__ Y) {
  const int i = blockIdx.x * 256 + threadIdx.x;   // grid sized exactly n/4
  const float4 v = ((const float4*)X)[i];
  v4h o;
  o[0] = (_Float16)v.x; o[1] = (_Float16)v.y;
  o[2] = (_Float16)v.z; o[3] = (_Float16)v.w;
  ((v4h*)Y)[i] = o;
}

__global__ void __launch_bounds__(256) wtrans_kernel(const float* __restrict__ W,
                                                     _Float16* __restrict__ Wt) {
  __shared__ float tile[32][33];
  const int bx = blockIdx.x * 32, by = blockIdx.y * 32;
  const int tx = threadIdx.x & 31, ty = threadIdx.x >> 5;
#pragma unroll
  for (int r = ty; r < 32; r += 8) tile[r][tx] = W[(size_t)(by + r) * D + bx + tx];
  __syncthreads();
#pragma unroll
  for (int r = ty; r < 32; r += 8)
    Wt[(size_t)(bx + r) * D + by + tx] = (_Float16)tile[tx][r];
}

// C/D layout: VGPR j holds M = j + 8*(lane>=16), N = lane%16.
template <int MODE>   // 0/1: [B,H,S,dh]   2: [B,H,dh,S]
__device__ __forceinline__ void store_tile_f16(v8f c, int tm, int tn, int lane,
                                               const float* __restrict__ bias,
                                               _Float16* __restrict__ Y) {
  const int n  = tn * 16 + (lane & 15);
  const float bv = bias[n];
  const int h = n >> 6, d = n & 63;
  const int mhi = (lane >> 4) << 3;
#pragma unroll
  for (int j = 0; j < 8; ++j) {
    const int m  = tm * 16 + mhi + j;
    const int bb = m >> 11;
    const int s  = m & (S - 1);
    const size_t idx = (MODE == 2)
        ? ((size_t)((bb * H + h) * DH + d)) * S + s
        : ((size_t)((bb * H + h) * S + s)) * DH + d;
    Y[idx] = (_Float16)(c[j] + bv);
  }
}

// ---- Y = Xh @ Wt^T + bias; 4 n-tiles/wave, software-pipelined loads ----
template <int MODE>
__global__ void __launch_bounds__(256) proj_kernel(const _Float16* __restrict__ X,
                                                   const _Float16* __restrict__ Wt,
                                                   const float* __restrict__ bias,
                                                   _Float16* __restrict__ Y) {
  const int lane = threadIdx.x & 31;
  const int wid  = threadIdx.x >> 5;
  const int tm   = blockIdx.x * 8 + wid;
  const int ng   = blockIdx.y;
  const int kb   = (lane >> 4) << 3;

  const _Float16* ar = X  + (size_t)(tm * 16 + (lane & 15)) * D + kb;
  const _Float16* b0 = Wt + (size_t)(ng * 64 +  0 + (lane & 15)) * D + kb;
  const _Float16* b1 = Wt + (size_t)(ng * 64 + 16 + (lane & 15)) * D + kb;
  const _Float16* b2 = Wt + (size_t)(ng * 64 + 32 + (lane & 15)) * D + kb;
  const _Float16* b3 = Wt + (size_t)(ng * 64 + 48 + (lane & 15)) * D + kb;

  v16h a  = load_frag(ar);
  v16h f0 = load_frag(b0), f1 = load_frag(b1);
  v16h f2 = load_frag(b2), f3 = load_frag(b3);
  v8f c0 = {}, c1 = {}, c2 = {}, c3 = {};
  for (int kk = 32; kk < D; kk += 32) {
    // issue next iteration's loads before this iteration's WMMAs
    const v16h an = load_frag(ar + kk);
    const v16h g0 = load_frag(b0 + kk);
    const v16h g1 = load_frag(b1 + kk);
    const v16h g2 = load_frag(b2 + kk);
    const v16h g3 = load_frag(b3 + kk);
    c0 = wmma_f16(a, f0, c0);
    c1 = wmma_f16(a, f1, c1);
    c2 = wmma_f16(a, f2, c2);
    c3 = wmma_f16(a, f3, c3);
    a = an; f0 = g0; f1 = g1; f2 = g2; f3 = g3;
  }
  c0 = wmma_f16(a, f0, c0);
  c1 = wmma_f16(a, f1, c1);
  c2 = wmma_f16(a, f2, c2);
  c3 = wmma_f16(a, f3, c3);

  store_tile_f16<MODE>(c0, tm, ng * 4 + 0, lane, bias, Y);
  store_tile_f16<MODE>(c1, tm, ng * 4 + 1, lane, bias, Y);
  store_tile_f16<MODE>(c2, tm, ng * 4 + 2, lane, bias, Y);
  store_tile_f16<MODE>(c3, tm, ng * 4 + 3, lane, bias, Y);
}

// ---- out = Ctx @ WoT^T + bo, f32 out [M][N] ----
__global__ void __launch_bounds__(256) out_proj_kernel(const _Float16* __restrict__ A,
                                                       const _Float16* __restrict__ Wt,
                                                       const float* __restrict__ bias,
                                                       float* __restrict__ Y) {
  const int lane = threadIdx.x & 31;
  const int wid  = threadIdx.x >> 5;
  const int tm   = blockIdx.x * 8 + wid;
  const int ng   = blockIdx.y;
  const int kb   = (lane >> 4) << 3;

  const _Float16* ar = A  + (size_t)(tm * 16 + (lane & 15)) * D + kb;
  const _Float16* b0 = Wt + (size_t)(ng * 64 +  0 + (lane & 15)) * D + kb;
  const _Float16* b1 = Wt + (size_t)(ng * 64 + 16 + (lane & 15)) * D + kb;
  const _Float16* b2 = Wt + (size_t)(ng * 64 + 32 + (lane & 15)) * D + kb;
  const _Float16* b3 = Wt + (size_t)(ng * 64 + 48 + (lane & 15)) * D + kb;

  v16h a  = load_frag(ar);
  v16h f0 = load_frag(b0), f1 = load_frag(b1);
  v16h f2 = load_frag(b2), f3 = load_frag(b3);
  v8f c0 = {}, c1 = {}, c2 = {}, c3 = {};
  for (int kk = 32; kk < D; kk += 32) {
    const v16h an = load_frag(ar + kk);
    const v16h g0 = load_frag(b0 + kk);
    const v16h g1 = load_frag(b1 + kk);
    const v16h g2 = load_frag(b2 + kk);
    const v16h g3 = load_frag(b3 + kk);
    c0 = wmma_f16(a, f0, c0);
    c1 = wmma_f16(a, f1, c1);
    c2 = wmma_f16(a, f2, c2);
    c3 = wmma_f16(a, f3, c3);
    a = an; f0 = g0; f1 = g1; f2 = g2; f3 = g3;
  }
  c0 = wmma_f16(a, f0, c0);
  c1 = wmma_f16(a, f1, c1);
  c2 = wmma_f16(a, f2, c2);
  c3 = wmma_f16(a, f3, c3);

  const int mhi = (lane >> 4) << 3;
  v8f cs[4] = {c0, c1, c2, c3};
#pragma unroll
  for (int t = 0; t < 4; ++t) {
    const int n = (ng * 4 + t) * 16 + (lane & 15);
    const float bv = bias[n];
#pragma unroll
    for (int j = 0; j < 8; ++j) {
      const int m = tm * 16 + mhi + j;
      Y[(size_t)m * D + n] = cs[t][j] + bv;
    }
  }
}

// ---- attention: one (b,h, 16-row q tile) per 256-thread workgroup ----
__global__ void __launch_bounds__(256) attn_kernel(const _Float16* __restrict__ Qh,
                                                   const _Float16* __restrict__ Kh,
                                                   const _Float16* __restrict__ VhT,
                                                   float* __restrict__ Wout,
                                                   _Float16* __restrict__ Ctx) {
  __shared__ float    sc[16][S];            // 128 KB raw scores
  __shared__ _Float16 ph16[16][S];          //  64 KB normalized P in f16
  __shared__ _Float16 kstage[8][2][16][DH]; //  32 KB async K double-buffers
  __shared__ float    red[16][17];
  __shared__ float    rowmax[16];
  __shared__ float    rowsum[16];
  __shared__ float    ctxacc[2][16][DH];

  const int lane = threadIdx.x & 31;
  const int wid  = threadIdx.x >> 5;
  const int tid  = threadIdx.x;
  const int kb   = (lane >> 4) << 3;

  const int qt = blockIdx.x & (S / 16 - 1);
  const int bh = blockIdx.x / (S / 16);
  const int b0 = bh / H;
  const int h0 = bh % H;

  const _Float16* Qb = Qh  + (size_t)bh * S * DH;
  const _Float16* Kb = Kh  + (size_t)bh * S * DH;
  const _Float16* Vb = VhT + (size_t)bh * DH * S;

  // phase 1: scores = (Q @ K^T)/8 with async double-buffered K tiles
  {
    const _Float16* qr = Qb + (size_t)(qt * 16 + (lane & 15)) * DH + kb;
    const v16h a0 = load_frag(qr);
    const v16h a1 = load_frag(qr + 32);

    _Float16* lb0 = &kstage[wid][0][lane & 15][kb];
    _Float16* lb1 = &kstage[wid][1][lane & 15][kb];
    // per-lane global src for tile: row = lane%16, 16B chunk = (lane/16)
    const _Float16* g0 = Kb + (size_t)(wid * 256 + (lane & 15)) * DH + kb;
    async_stage_tile(g0, lb0);                         // tile 0 in flight

    const int mhi = (lane >> 4) << 3;
    for (int t = 0; t < 15; ++t) {
      const _Float16* cur = (t & 1) ? lb1 : lb0;
      _Float16*       nxt = (t & 1) ? lb0 : lb1;
      const _Float16* gn =
          Kb + (size_t)((wid * 16 + t + 1) * 16 + (lane & 15)) * DH + kb;
      async_stage_tile(gn, nxt);                       // prefetch t+1
      asm volatile("s_wait_asynccnt 0x4" ::: "memory"); // tile t resident
      v8f c = {};
      c = wmma_f16(a0, load_frag(cur), c);              // ds_load_b128 frags
      c = wmma_f16(a1, load_frag(cur + 32), c);
      const int col = (wid * 16 + t) * 16 + (lane & 15);
#pragma unroll
      for (int j = 0; j < 8; ++j) sc[mhi + j][col] = c[j] * 0.125f;
    }
    { // last tile (no further prefetch)
      asm volatile("s_wait_asynccnt 0x0" ::: "memory");
      const _Float16* cur = lb1;                        // t = 15 -> buf 1
      v8f c = {};
      c = wmma_f16(a0, load_frag(cur), c);
      c = wmma_f16(a1, load_frag(cur + 32), c);
      const int col = (wid * 16 + 15) * 16 + (lane & 15);
#pragma unroll
      for (int j = 0; j < 8; ++j) sc[mhi + j][col] = c[j] * 0.125f;
    }
  }
  __syncthreads();

  // phase 2: exact softmax across 2048-wide rows (float4 scans)
  const int r     = tid & 15;
  const int chunk = tid >> 4;            // 16 chunks x 128 cols
  {
    const float4* row = (const float4*)&sc[r][chunk * 128];
    float m = -3.4e38f;
    for (int i = 0; i < 32; ++i) {
      const float4 v = row[i];
      m = fmaxf(m, fmaxf(fmaxf(v.x, v.y), fmaxf(v.z, v.w)));
    }
    red[r][chunk] = m;
  }
  __syncthreads();
  if (tid < 16) {
    float m = red[tid][0];
    for (int i = 1; i < 16; ++i) m = fmaxf(m, red[tid][i]);
    rowmax[tid] = m;
  }
  __syncthreads();
  {
    const float m = rowmax[r];
    float4* row = (float4*)&sc[r][chunk * 128];
    float s = 0.f;
    for (int i = 0; i < 32; ++i) {
      float4 v = row[i];
      v.x = __expf(v.x - m); v.y = __expf(v.y - m);
      v.z = __expf(v.z - m); v.w = __expf(v.w - m);
      row[i] = v;
      s += v.x + v.y + v.z + v.w;
    }
    red[r][chunk] = s;
  }
  __syncthreads();
  if (tid < 16) {
    float s = red[tid][0];
    for (int i = 1; i < 16; ++i) s += red[tid][i];
    rowsum[tid] = s;
  }
  __syncthreads();
  {
    // normalize: stream f32 weights to HBM (dominant 1.07GB store) and keep a
    // f16 copy of P in LDS so phase 3 A-frags need zero conversions.
    const float inv = 1.0f / rowsum[r];
    const float4* row = (const float4*)&sc[r][chunk * 128];
    float4* g  = (float4*)(Wout + ((size_t)bh * S + qt * 16 + r) * S + chunk * 128);
    v4h* p16 = (v4h*)&ph16[r][chunk * 128];
    for (int i = 0; i < 32; ++i) {
      float4 v = row[i];
      v.x *= inv; v.y *= inv; v.z *= inv; v.w *= inv;
      g[i] = v;                               // global_store_b128 stream
      v4h hv;
      hv[0] = (_Float16)v.x; hv[1] = (_Float16)v.y;
      hv[2] = (_Float16)v.z; hv[3] = (_Float16)v.w;
      p16[i] = hv;
    }
  }
  __syncthreads();

  // phase 3: ctx = P @ V; 8 waves = 4 d-tiles x 2 K-halves, pipelined
  {
    const int dtile = wid & 3;
    const int khalf = wid >> 2;
    const _Float16* vr = Vb + (size_t)(dtile * 16 + (lane & 15)) * S + kb;
    const _Float16* pr = &ph16[lane & 15][kb];
    const int kbase = khalf * 1024;

    v16h aC = load_frag(pr + kbase);
    v16h bC = load_frag(vr + kbase);
    v8f c = {};
    for (int ks = 1; ks < 32; ++ks) {
      const int kk = kbase + ks * 32;
      const v16h aN = load_frag(pr + kk);   // ds_load_b128
      const v16h bN = load_frag(vr + kk);   // global_load_b128
      c = wmma_f16(aC, bC, c);
      aC = aN; bC = bN;
    }
    c = wmma_f16(aC, bC, c);

    const int dcol = dtile * 16 + (lane & 15);
    const int mhi  = (lane >> 4) << 3;
#pragma unroll
    for (int j = 0; j < 8; ++j) ctxacc[khalf][mhi + j][dcol] = c[j];
  }
  __syncthreads();

  // phase 4: combine split-K halves, store ctx (f16) in [B,S,D]
  for (int e = tid; e < 16 * DH; e += 256) {
    const int mr = e >> 6, d = e & 63;
    const float v = ctxacc[0][mr][d] + ctxacc[1][mr][d];
    Ctx[((size_t)(b0 * S + qt * 16 + mr)) * D + h0 * DH + d] = (_Float16)v;
  }
}

extern "C" void kernel_launch(void* const* d_in, const int* in_sizes, int n_in,
                              void* d_out, int out_size, void* d_ws, size_t ws_size,
                              hipStream_t stream) {
  (void)in_sizes; (void)n_in; (void)out_size; (void)ws_size;
  const float* q  = (const float*)d_in[0];
  const float* k  = (const float*)d_in[1];
  const float* v  = (const float*)d_in[2];
  const float* Wq = (const float*)d_in[3];
  const float* bq = (const float*)d_in[4];
  const float* Wk = (const float*)d_in[5];
  const float* bk = (const float*)d_in[6];
  const float* Wv = (const float*)d_in[7];
  const float* bv = (const float*)d_in[8];
  const float* Wo = (const float*)d_in[9];
  const float* bo = (const float*)d_in[10];

  float* out = (float*)d_out;
  float* wts = out + (size_t)B * S * D;       // outputs concat: out, weights

  const size_t per = (size_t)B * H * S * DH;  // 8.39M halves
  _Float16* Qh  = (_Float16*)d_ws;
  _Float16* Kh  = Qh  + per;
  _Float16* VhT = Kh  + per;
  _Float16* Ctx = VhT + per;
  _Float16* qh16 = Ctx + per;
  _Float16* kh16 = qh16 + per;
  _Float16* vh16 = kh16 + per;
  _Float16* WqT = vh16 + per;                 // f16 transposed weights [N][K]
  _Float16* WkT = WqT + (size_t)D * D;
  _Float16* WvT = WkT + (size_t)D * D;
  _Float16* WoT = WvT + (size_t)D * D;

  const int nconv = (int)(per / 4);
  cvt16_kernel<<<nconv / 256, 256, 0, stream>>>(q, qh16);
  cvt16_kernel<<<nconv / 256, 256, 0, stream>>>(k, kh16);
  cvt16_kernel<<<nconv / 256, 256, 0, stream>>>(v, vh16);
  dim3 gt(D / 32, D / 32);
  wtrans_kernel<<<gt, 256, 0, stream>>>(Wq, WqT);
  wtrans_kernel<<<gt, 256, 0, stream>>>(Wk, WkT);
  wtrans_kernel<<<gt, 256, 0, stream>>>(Wv, WvT);
  wtrans_kernel<<<gt, 256, 0, stream>>>(Wo, WoT);

  dim3 gproj(64, 16);
  proj_kernel<0><<<gproj, 256, 0, stream>>>(qh16, WqT, bq, Qh);
  proj_kernel<1><<<gproj, 256, 0, stream>>>(kh16, WkT, bk, Kh);
  proj_kernel<2><<<gproj, 256, 0, stream>>>(vh16, WvT, bv, VhT);

  attn_kernel<<<B * H * (S / 16), 256, 0, stream>>>(Qh, Kh, VhT, wts, Ctx);
  out_proj_kernel<<<gproj, 256, 0, stream>>>(Ctx, WoT, bo, out);
}
